// AttentiveEncoderDecoderNetwork_43422119362843
// MI455X (gfx1250) — matrix operand order repaired and checked by reading, not compiled, not run on hardware
//
#include <hip/hip_runtime.h>
#include <hip/hip_bf16.h>
#include <math.h>

// ---------------------------------------------------------------------------
// Types for CDNA5 WMMA
// ---------------------------------------------------------------------------
typedef __attribute__((ext_vector_type(16))) __bf16 v16bf;
typedef __attribute__((ext_vector_type(8)))  __bf16 v8bf;
typedef __attribute__((ext_vector_type(8)))  float  v8f;

// A-fragment (16x32 bf16, MxK): lane = {half,row16}; per ISA layout:
//   lanes 0-15 : K = 0..7 and 16..23 ; lanes 16-31 : K = 8..15 and 24..31
// => per lane: contiguous [half*8 .. half*8+7] and [16+half*8 .. +7]
__device__ __forceinline__ v16bf frag_a(const __bf16* rowp, int half) {
  v8bf lo = *(const v8bf*)(rowp + half * 8);
  v8bf hi = *(const v8bf*)(rowp + 16 + half * 8);
  return __builtin_shufflevector(lo, hi, 0, 1, 2, 3, 4, 5, 6, 7,
                                 8, 9, 10, 11, 12, 13, 14, 15);
}
// B-fragment (32x16 bf16, KxN), LDS stored as [N][K]:
//   lanes 0-15 hold K=0..7,16..23, lanes 16-31 hold K=8..15,24..31
__device__ __forceinline__ v16bf frag_b(const __bf16* colp, int half) {
  v8bf lo = *(const v8bf*)(colp + half * 8);
  v8bf hi = *(const v8bf*)(colp + 16 + half * 8);
  return __builtin_shufflevector(lo, hi, 0, 1, 2, 3, 4, 5, 6, 7,
                                 8, 9, 10, 11, 12, 13, 14, 15);
}
__device__ __forceinline__ v8f wmma_bf16(v16bf a, v16bf b, v8f c) {
  return __builtin_amdgcn_wmma_f32_16x16x32_bf16(false, a, false, b,
                                                 (short)0, c, false, false);
}

__device__ __forceinline__ int refl(int i, int n) {
  if (i < 0) i = -i;
  if (i >= n) i = 2 * n - 2 - i;
  return i;
}

// ---------------------------------------------------------------------------
// Direct convolution with reflect padding (NCHW, OIHW), act: 0=ELU, 1=ReLU
// ---------------------------------------------------------------------------
template <int KS>
__global__ __launch_bounds__(256) void conv_refl(
    const float* __restrict__ in, const float* __restrict__ w,
    const float* __restrict__ bias, float* __restrict__ out,
    int Cin, int Cout, int Hin, int Win, int Hout, int Wout,
    int stride, int act) {
  int idx = blockIdx.x * 256 + threadIdx.x;
  int total = Cout * Hout * Wout;
  if (idx >= total) return;
  int ox = idx % Wout;
  int t = idx / Wout;
  int oy = t % Hout;
  int co = t / Hout;
  constexpr int PAD = KS / 2;
  int iy[KS], ix[KS];
#pragma unroll
  for (int k = 0; k < KS; k++) {
    iy[k] = refl(oy * stride + k - PAD, Hin);
    ix[k] = refl(ox * stride + k - PAD, Win);
  }
  float acc = bias[co];
  const float* wp = w + (size_t)co * Cin * KS * KS;
  for (int ci = 0; ci < Cin; ci++) {
    const float* ip = in + (size_t)ci * Hin * Win;
    const float* wc = wp + ci * KS * KS;
#pragma unroll
    for (int ky = 0; ky < KS; ky++) {
      const float* irow = ip + (size_t)iy[ky] * Win;
#pragma unroll
      for (int kx = 0; kx < KS; kx++)
        acc = fmaf(irow[ix[kx]], wc[ky * KS + kx], acc);
    }
  }
  out[idx] = act ? fmaxf(acc, 0.f) : (acc > 0.f ? acc : expm1f(acc));
}

// ---------------------------------------------------------------------------
// Build patch matrices (bf16, transposed: [p=9216][k=1152]) from h6 + mask.
// fg = downsample(h6) * downsample(mask); bg = downsample(h6); zero pad 1.
// ---------------------------------------------------------------------------
__global__ __launch_bounds__(256) void prep_patches(
    const float* __restrict__ h6, const float* __restrict__ mask,
    __bf16* __restrict__ fgpT, __bf16* __restrict__ bgpT) {
  int idx = blockIdx.x * 256 + threadIdx.x;
  if (idx >= 9216 * 128) return;
  int c = idx & 127;
  int p = idx >> 7;
  int y = p / 96, x = p % 96;
#pragma unroll
  for (int dy = 0; dy < 3; dy++)
#pragma unroll
    for (int dx = 0; dx < 3; dx++) {
      int yy = y + dy - 1, xx = x + dx - 1;
      float vb = 0.f, vf = 0.f;
      if (yy >= 0 && yy < 96 && xx >= 0 && xx < 96) {
        float tv = h6[(size_t)c * 36864 + (2 * yy) * 192 + 2 * xx];
        float mm = mask[(2 * yy) * 192 + 2 * xx];
        vb = tv;
        vf = tv * mm;
      }
      int k = c * 9 + dy * 3 + dx;
      fgpT[(size_t)p * 1152 + k] = (__bf16)vf;
      bgpT[(size_t)p * 1152 + k] = (__bf16)vb;
    }
}

// ---------------------------------------------------------------------------
// Pass 1: S = fgpT * bgpT^T tilewise via WMMA; emit per-block sum(S^2).
// Grid (72, 72), 256 threads (8 waves, 4x2 wave grid; wave = 32x64 of 128x128)
// Double-buffered LDS staging: one barrier per k-step, global loads for the
// next tile issued before the current tile's WMMAs.
// ---------------------------------------------------------------------------
__global__ __launch_bounds__(256) void attn_pass1(
    const __bf16* __restrict__ fgpT, const __bf16* __restrict__ bgpT,
    float* __restrict__ partials) {
  __shared__ __bf16 As[2][128][40];
  __shared__ __bf16 Bs[2][128][40];
  const int tid = threadIdx.x;
  const int lane = tid & 31, wid = tid >> 5;
  const int half = lane >> 4, l16 = lane & 15;
  const int wr = wid >> 1, wc = wid & 1;
  const int row0 = blockIdx.x * 128, col0 = blockIdx.y * 128;
  const int r = tid >> 1, seg = (tid & 1) * 16;
  const __bf16* aSrc = fgpT + (size_t)(row0 + r) * 1152 + seg;
  const __bf16* bSrc = bgpT + (size_t)(col0 + r) * 1152 + seg;

  v8f acc[2][4] = {};
  uint4 ra0 = *(const uint4*)(aSrc);
  uint4 ra1 = *(const uint4*)(aSrc + 8);
  uint4 rb0 = *(const uint4*)(bSrc);
  uint4 rb1 = *(const uint4*)(bSrc + 8);
  int buf = 0;
  for (int k0 = 0; k0 < 1152; k0 += 32) {
    *(uint4*)&As[buf][r][seg] = ra0;
    *(uint4*)&As[buf][r][seg + 8] = ra1;
    *(uint4*)&Bs[buf][r][seg] = rb0;
    *(uint4*)&Bs[buf][r][seg + 8] = rb1;
    __syncthreads();
    if (k0 + 32 < 1152) {
      ra0 = *(const uint4*)(aSrc + k0 + 32);
      ra1 = *(const uint4*)(aSrc + k0 + 40);
      rb0 = *(const uint4*)(bSrc + k0 + 32);
      rb1 = *(const uint4*)(bSrc + k0 + 40);
    }
    v16bf af[2], bf4[4];
#pragma unroll
    for (int rt = 0; rt < 2; rt++)
      af[rt] = frag_a(&As[buf][wr * 32 + rt * 16 + l16][0], half);
#pragma unroll
    for (int ct = 0; ct < 4; ct++)
      bf4[ct] = frag_b(&Bs[buf][wc * 64 + ct * 16 + l16][0], half);
#pragma unroll
    for (int rt = 0; rt < 2; rt++)
#pragma unroll
      for (int ct = 0; ct < 4; ct++)
        acc[rt][ct] = wmma_bf16(af[rt], bf4[ct], acc[rt][ct]);
    buf ^= 1;
  }
  float loc = 0.f;
#pragma unroll
  for (int rt = 0; rt < 2; rt++)
#pragma unroll
    for (int ct = 0; ct < 4; ct++)
#pragma unroll
      for (int v = 0; v < 8; v++) loc += acc[rt][ct][v] * acc[rt][ct][v];
  for (int off = 16; off; off >>= 1) loc += __shfl_down(loc, off, 32);
  __shared__ float part[8];
  if (lane == 0) part[wid] = loc;
  __syncthreads();
  if (tid == 0) {
    float tsum = 0.f;
    for (int i = 0; i < 8; i++) tsum += part[i];
    partials[blockIdx.y * 72 + blockIdx.x] = tsum;
  }
}

// Deterministic reduction of the 72*72 partials into ss[0].
__global__ void reduce_ss(const float* __restrict__ partials,
                          float* __restrict__ ss) {
  if (blockIdx.x == 0 && threadIdx.x == 0) {
    float s = 0.f;
    for (int i = 0; i < 72 * 72; i++) s += partials[i];
    ss[0] = s;
  }
}

// ---------------------------------------------------------------------------
// Pass 2: flash-style. Grid (72 row-tiles, 18 channel-blocks of 64).
// Per q-tile: recompute S (WMMA, double-buffered LDS), P = exp(scale*S)
// (|arg| <= 10 by the Frobenius bound -> no max subtraction needed),
// R_acc += P * bgpT-block (WMMA).  cBlock 0 also emits rowSum.
// ---------------------------------------------------------------------------
__global__ __launch_bounds__(256) void attn_pass2(
    const __bf16* __restrict__ fgpT, const __bf16* __restrict__ bgpT,
    const float* __restrict__ ss, float* __restrict__ Rm,
    float* __restrict__ rowSum) {
  __shared__ __bf16 As[2][128][40];
  __shared__ __bf16 Bs[2][128][40];
  __shared__ __bf16 Ps[128][136];
  __shared__ __bf16 Bt2[64][136];
  const int tid = threadIdx.x;
  const int lane = tid & 31, wid = tid >> 5;
  const int half = lane >> 4, l16 = lane & 15;
  const int wr = wid >> 1, wc = wid & 1;
  const int row0 = blockIdx.x * 128;
  const int c0 = blockIdx.y * 64;
  const float scale = 10.f / fmaxf(sqrtf(ss[0]), 1e-12f);
  const int r = tid >> 1, seg = (tid & 1) * 16;
  const __bf16* aSrc = fgpT + (size_t)(row0 + r) * 1152 + seg;

  v8f racc[2][2] = {};
  float rsum = 0.f;

  for (int q0 = 0; q0 < 9216; q0 += 128) {
    v8f sacc[2][4] = {};
    const __bf16* bSrc = bgpT + (size_t)(q0 + r) * 1152 + seg;
    uint4 ra0 = *(const uint4*)(aSrc);
    uint4 ra1 = *(const uint4*)(aSrc + 8);
    uint4 rb0 = *(const uint4*)(bSrc);
    uint4 rb1 = *(const uint4*)(bSrc + 8);
    int buf = 0;
    for (int k0 = 0; k0 < 1152; k0 += 32) {
      *(uint4*)&As[buf][r][seg] = ra0;
      *(uint4*)&As[buf][r][seg + 8] = ra1;
      *(uint4*)&Bs[buf][r][seg] = rb0;
      *(uint4*)&Bs[buf][r][seg + 8] = rb1;
      __syncthreads();
      if (k0 + 32 < 1152) {
        ra0 = *(const uint4*)(aSrc + k0 + 32);
        ra1 = *(const uint4*)(aSrc + k0 + 40);
        rb0 = *(const uint4*)(bSrc + k0 + 32);
        rb1 = *(const uint4*)(bSrc + k0 + 40);
      }
      v16bf af[2], bf4[4];
#pragma unroll
      for (int rt = 0; rt < 2; rt++)
        af[rt] = frag_a(&As[buf][wr * 32 + rt * 16 + l16][0], half);
#pragma unroll
      for (int ct = 0; ct < 4; ct++)
        bf4[ct] = frag_b(&Bs[buf][wc * 64 + ct * 16 + l16][0], half);
#pragma unroll
      for (int rt = 0; rt < 2; rt++)
#pragma unroll
        for (int ct = 0; ct < 4; ct++)
          sacc[rt][ct] = wmma_bf16(af[rt], bf4[ct], sacc[rt][ct]);
      buf ^= 1;
    }
    __syncthreads();
    // P = exp(scale * S)
#pragma unroll
    for (int rt = 0; rt < 2; rt++)
#pragma unroll
      for (int ct = 0; ct < 4; ct++)
#pragma unroll
        for (int v = 0; v < 8; v++) {
          int rr = wr * 32 + rt * 16 + half * 8 + v;
          int cc = wc * 64 + ct * 16 + l16;
          Ps[rr][cc] = (__bf16)__expf(scale * sacc[rt][ct][v]);
        }
    __syncthreads();
    if (blockIdx.y == 0 && tid < 128) {
#pragma unroll
      for (int q = 0; q < 128; q += 8) {
        v8bf ch = *(const v8bf*)&Ps[tid][q];
#pragma unroll
        for (int j = 0; j < 8; j++) rsum += (float)ch[j];
      }
    }
    // Stage Bt2[c][q] (transposed) from bgpT[q0+q][c0+c]
    {
      int q = tid >> 1;
      int cseg = (tid & 1) * 32;
      const uint4* src =
          (const uint4*)(bgpT + (size_t)(q0 + q) * 1152 + c0 + cseg);
      __bf16 tmp[32];
      *(uint4*)&tmp[0] = src[0];
      *(uint4*)&tmp[8] = src[1];
      *(uint4*)&tmp[16] = src[2];
      *(uint4*)&tmp[24] = src[3];
#pragma unroll
      for (int i = 0; i < 32; i++) Bt2[cseg + i][q] = tmp[i];
    }
    __syncthreads();
    // R_acc += P(128x128) x bgpT-block(128x64)
#pragma unroll
    for (int kq = 0; kq < 128; kq += 32) {
      v16bf pa[2], pb[2];
#pragma unroll
      for (int rt = 0; rt < 2; rt++)
        pa[rt] = frag_a(&Ps[wr * 32 + rt * 16 + l16][kq], half);
#pragma unroll
      for (int ct = 0; ct < 2; ct++)
        pb[ct] = frag_b(&Bt2[wc * 32 + ct * 16 + l16][kq], half);
#pragma unroll
      for (int rt = 0; rt < 2; rt++)
#pragma unroll
        for (int ct = 0; ct < 2; ct++)
          racc[rt][ct] = wmma_bf16(pa[rt], pb[ct], racc[rt][ct]);
    }
    __syncthreads();
  }
#pragma unroll
  for (int rt = 0; rt < 2; rt++)
#pragma unroll
    for (int ct = 0; ct < 2; ct++)
#pragma unroll
      for (int v = 0; v < 8; v++) {
        int rr = row0 + wr * 32 + rt * 16 + half * 8 + v;
        int cc = c0 + wc * 32 + ct * 16 + l16;
        Rm[(size_t)rr * 1152 + cc] = racc[rt][ct][v];
      }
  if (blockIdx.y == 0 && tid < 128) rowSum[row0 + tid] = rsum;
}

// ---------------------------------------------------------------------------
// Fold 3x3 patches back, crop, normalize by softmax denominator, mask, 2x up.
// ---------------------------------------------------------------------------
__global__ __launch_bounds__(256) void fold_upsample(
    const float* __restrict__ Rm, const float* __restrict__ rowSum,
    const float* __restrict__ mask, float* __restrict__ ctx) {
  int idx = blockIdx.x * 256 + threadIdx.x;
  if (idx >= 128 * 96 * 96) return;
  int x = idx % 96;
  int t = idx / 96;
  int y = t % 96;
  int c = t / 96;
  float s = 0.f;
#pragma unroll
  for (int dy = 0; dy < 3; dy++)
#pragma unroll
    for (int dx = 0; dx < 3; dx++) {
      int yy = y + 1 - dy, xx = x + 1 - dx;
      if (yy >= 0 && yy < 96 && xx >= 0 && xx < 96) {
        int p = yy * 96 + xx;
        s += Rm[(size_t)p * 1152 + c * 9 + dy * 3 + dx] / rowSum[p];
      }
    }
  s *= mask[(2 * y) * 192 + 2 * x];
  float* o = ctx + (size_t)c * 36864 + (2 * y) * 192 + 2 * x;
  o[0] = s;
  o[1] = s;
  o[192] = s;
  o[193] = s;
}

// ---------------------------------------------------------------------------
// Host side
// ---------------------------------------------------------------------------
extern "C" void kernel_launch(void* const* d_in, const int* in_sizes, int n_in,
                              void* d_out, int out_size, void* d_ws,
                              size_t ws_size, hipStream_t stream) {
  (void)in_sizes; (void)n_in; (void)out_size; (void)ws_size;
  const float* x = (const float*)d_in[0];
  const float* mask = (const float*)d_in[1];
  const float* w[8];
  const float* b[8];
  for (int i = 0; i < 8; i++) {
    w[i] = (const float*)d_in[2 + 2 * i];
    b[i] = (const float*)d_in[3 + 2 * i];
  }
  char* ws = (char*)d_ws;
  size_t off = 0;
  auto take = [&](size_t bytes) -> char* {
    char* p = ws + off;
    off += (bytes + 255) & ~(size_t)255;
    return p;
  };
  float* h1 = (float*)take(32ull * 768 * 768 * 4);
  float* h2 = (float*)take(32ull * 384 * 384 * 4);
  float* h3 = (float*)take(64ull * 384 * 384 * 4);
  float* h4 = (float*)take(128ull * 192 * 192 * 4);
  float* h5 = (float*)take(128ull * 192 * 192 * 4);
  float* h6 = (float*)take(128ull * 192 * 192 * 4);
  __bf16* fgpT = (__bf16*)take(9216ull * 1152 * 2);
  __bf16* bgpT = (__bf16*)take(9216ull * 1152 * 2);
  float* partials = (float*)take(72ull * 72 * 4);
  float* ssum = (float*)take(256);
  float* Rm = (float*)take(9216ull * 1152 * 4);
  float* rowSum = (float*)take(9216ull * 4);
  float* ctx = (float*)take(128ull * 192 * 192 * 4);
  float* h7 = (float*)take(128ull * 192 * 192 * 4);

  auto blocks = [](int total) { return (total + 255) / 256; };

  conv_refl<5><<<blocks(32 * 768 * 768), 256, 0, stream>>>(
      x, w[0], b[0], h1, 3, 32, 768, 768, 768, 768, 1, 0);
  conv_refl<3><<<blocks(32 * 384 * 384), 256, 0, stream>>>(
      h1, w[1], b[1], h2, 32, 32, 768, 768, 384, 384, 2, 0);
  conv_refl<3><<<blocks(64 * 384 * 384), 256, 0, stream>>>(
      h2, w[2], b[2], h3, 32, 64, 384, 384, 384, 384, 1, 0);
  conv_refl<3><<<blocks(128 * 192 * 192), 256, 0, stream>>>(
      h3, w[3], b[3], h4, 64, 128, 384, 384, 192, 192, 2, 0);
  conv_refl<3><<<blocks(128 * 192 * 192), 256, 0, stream>>>(
      h4, w[4], b[4], h5, 128, 128, 192, 192, 192, 192, 1, 0);
  conv_refl<3><<<blocks(128 * 192 * 192), 256, 0, stream>>>(
      h5, w[5], b[5], h6, 128, 128, 192, 192, 192, 192, 1, 1);

  prep_patches<<<blocks(9216 * 128), 256, 0, stream>>>(h6, mask, fgpT, bgpT);
  attn_pass1<<<dim3(72, 72), 256, 0, stream>>>(fgpT, bgpT, partials);
  reduce_ss<<<1, 1, 0, stream>>>(partials, ssum);
  attn_pass2<<<dim3(72, 18), 256, 0, stream>>>(fgpT, bgpT, ssum, Rm, rowSum);
  fold_upsample<<<blocks(128 * 96 * 96), 256, 0, stream>>>(Rm, rowSum, mask,
                                                           ctx);

  conv_refl<3><<<blocks(128 * 192 * 192), 256, 0, stream>>>(
      ctx, w[6], b[6], h7, 128, 128, 192, 192, 192, 192, 1, 0);
  conv_refl<3><<<blocks(128 * 192 * 192), 256, 0, stream>>>(
      h7, w[7], b[7], (float*)d_out, 128, 128, 192, 192, 192, 192, 1, 0);
}
